// SynthesisBlock_60705067762164
// MI455X (gfx1250) — matrix operand order repaired and checked by reading, not compiled, not run on hardware
//
#include <hip/hip_runtime.h>
#include <hip/hip_bf16.h>
#include <stdint.h>

typedef __attribute__((ext_vector_type(16))) __bf16 v16bf;
typedef __attribute__((ext_vector_type(8)))  __bf16 v8bf;
typedef __attribute__((ext_vector_type(8)))  float  v8f;
typedef int v4i __attribute__((vector_size(16)));
typedef __attribute__((address_space(1))) v4i* gv4i_p;   // global (AS1) int4*
typedef __attribute__((address_space(3))) v4i* lv4i_p;   // LDS    (AS3) int4*

#define B_    8
#define CIN   512
#define COUT  512
#define SDIM  512
#define HI    32
#define WI    32
#define HO    64
#define WO    64
#define NPIX  (HO*WO)
#define LRELU 0.2f
#define EPS_  1e-8f

// conv tiling
#define BM      64                  // couts per block
#define ROWS    2                   // output rows per block
#define INROWS  (ROWS+2)            // input rows incl. halo
#define KC      32                  // cin per K-chunk (one bf16 WMMA K)
#define NCHUNK  (CIN/KC)            // 16
#define IN_COLS 66                  // 64 + 1 halo each side
#define CSTR    40                  // padded cin stride in LDS (bank-spread, 16B aligned)
#define WELEMS  (9*BM*CSTR)         // 23040 bf16 per weight buffer
#define IELEMS  (INROWS*IN_COLS*CSTR) // 10560 bf16 per input buffer
#define LDS_BYTES ((size_t)2*(WELEMS+IELEMS)*2)   // double-buffered: 134,400 B

// ---- gfx1250 async global->LDS path (guarded; falls back to VGPR staging) ----
#if defined(__has_builtin)
#  if __has_builtin(__builtin_amdgcn_global_load_async_to_lds_b128)
#    define HAVE_ASYNC_LDS 1
#  endif
#  if __has_builtin(__builtin_amdgcn_s_wait_asynccnt)
#    define HAVE_WAIT_ASYNC 1
#  endif
#endif

__device__ __forceinline__ void cp16_async(__bf16* lds, const __bf16* g) {
#ifdef HAVE_ASYNC_LDS
  __builtin_amdgcn_global_load_async_to_lds_b128(
      (gv4i_p)(uintptr_t)g, (lv4i_p)(uintptr_t)lds, 0, 0);
#else
  *(v8bf*)lds = *(const v8bf*)g;
#endif
}

__device__ __forceinline__ void wait_async_lds() {
#ifdef HAVE_ASYNC_LDS
#  ifdef HAVE_WAIT_ASYNC
  __builtin_amdgcn_s_wait_asynccnt(0);
#  else
  asm volatile("s_wait_asynccnt 0x0" ::: "memory");
#  endif
#endif
}

// ---------------------------------------------------------------------------
// s[b,c] = w[b,:] . mod_w[c,:] + mod_b[c]
__global__ void style_kernel(const float* __restrict__ w,
                             const float* __restrict__ mod_w,
                             const float* __restrict__ mod_b,
                             float* __restrict__ s) {
  int c = blockIdx.x * blockDim.x + threadIdx.x;
  int b = blockIdx.y;
  if (c >= CIN) return;
  float acc = mod_b[c];
  const float* wr = w + b * SDIM;
  const float* mr = mod_w + (size_t)c * SDIM;
  for (int k = 0; k < SDIM; ++k) acc += wr[k] * mr[k];
  s[b * CIN + c] = acc;
}

// wsqT[cin, cout] = sum_{k=0..8} W[cout,cin,k]^2   (transposed for coalesced demod)
__global__ void wsqT_kernel(const float* __restrict__ wgt, float* __restrict__ wsqT) {
  int i = blockIdx.x * blockDim.x + threadIdx.x;
  if (i >= COUT * CIN) return;
  int cin = i % CIN, cout = i / CIN;
  const float* p = wgt + (size_t)i * 9;
  float a = 0.f;
#pragma unroll
  for (int t = 0; t < 9; ++t) a += p[t] * p[t];
  wsqT[(size_t)cin * COUT + cout] = a;
}

// d[b,cout] = rsqrt( sum_cin s[b,cin]^2 * wsqT[cin,cout] + eps )
__global__ void demod_kernel(const float* __restrict__ s,
                             const float* __restrict__ wsqT,
                             float* __restrict__ d) {
  int cout = blockIdx.x * blockDim.x + threadIdx.x;
  int b = blockIdx.y;
  if (cout >= COUT) return;
  float a = 0.f;
  for (int cin = 0; cin < CIN; ++cin) {
    float sv = s[b * CIN + cin];
    a += sv * sv * wsqT[(size_t)cin * COUT + cout];
  }
  d[b * COUT + cout] = rsqrtf(a + EPS_);
}

// wrep[tap][cout][cin] (bf16) <- weight[cout][cin][ky][kx]
__global__ void repack_kernel(const float* __restrict__ wgt, __bf16* __restrict__ wrep) {
  size_t i = blockIdx.x * (size_t)blockDim.x + threadIdx.x;
  if (i >= (size_t)9 * COUT * CIN) return;
  int cin  = (int)(i % CIN);
  int cout = (int)((i / CIN) % COUT);
  int tap  = (int)(i / ((size_t)CIN * COUT));
  float v = wgt[((size_t)cout * CIN + cin) * 9 + tap];
  wrep[((size_t)tap * COUT + cout) * CIN + cin] = (__bf16)v;
}

// up[b,Y,X,c] (bf16, NHWC) = bilinear2x(x[b,c])(Y,X) * s1[b,c]
__global__ __launch_bounds__(256)
void upmod_nhwc_kernel(const float* __restrict__ x,
                       const float* __restrict__ s1,
                       __bf16* __restrict__ up) {
  __shared__ float xs[2][32][33];   // [row][c][x], padded to kill bank conflicts
  const int tid = threadIdx.x;
  const int b = blockIdx.z, Y = blockIdx.y, cc = blockIdx.x * 32;

  float fy = Y * 0.5f - 0.25f;
  int y0 = (int)floorf(fy); float wy = fy - (float)y0;
  int y0c = min(max(y0, 0), HI - 1), y1c = min(max(y0 + 1, 0), HI - 1);

  for (int i = tid; i < 2 * 32 * 32; i += 256) {
    int xx = i & 31, c = (i >> 5) & 31, rr = i >> 10;
    int yrow = rr ? y1c : y0c;
    xs[rr][c][xx] = x[(((size_t)b * CIN + cc + c) * HI + yrow) * WI + xx];
  }
  __syncthreads();
  for (int i = tid; i < WO * 32; i += 256) {
    int c = i & 31, X = i >> 5;
    float fx = X * 0.5f - 0.25f;
    int x0 = (int)floorf(fx); float wx = fx - (float)x0;
    int x0c = min(max(x0, 0), WI - 1), x1c = min(max(x0 + 1, 0), WI - 1);
    float v0 = (1.f - wx) * xs[0][c][x0c] + wx * xs[0][c][x1c];
    float v1 = (1.f - wx) * xs[1][c][x0c] + wx * xs[1][c][x1c];
    float v = (1.f - wy) * v0 + wy * v1;
    up[(((size_t)b * HO + Y) * WO + X) * CIN + cc + c] = (__bf16)(v * s1[b * CIN + cc + c]);
  }
}

// ---------------------------------------------------------------------------
// Implicit-GEMM 3x3 conv via v_wmma_f32_16x16x32_bf16.
// 128 threads (4 wave32). Block tile = BM(64) couts x ROWS(2) rows x 64 px.
// Double-buffered LDS; global->LDS via async copies (prefetch next K-chunk
// while computing the current one). Each A fragment feeds 8 WMMAs.
__global__ __launch_bounds__(128, 1)
void conv3x3_wmma(const __bf16* __restrict__ in,      // [B][HO][WO][CIN]  NHWC
                  const __bf16* __restrict__ wrep,    // [9][COUT][CIN]
                  const float*  __restrict__ dmod,    // [B][COUT]
                  const float*  __restrict__ noise,   // [B][64][64]
                  const float*  __restrict__ nscale,  // [COUT]
                  const float*  __restrict__ postsc,  // [B][COUT] or null (next conv's modulation)
                  float*        __restrict__ out_f32, // [B][COUT][64][64] NCHW or null
                  __bf16*       __restrict__ out_bf16)// [B][HO][WO][COUT] NHWC or null
{
  extern __shared__ __bf16 smem[];   // [ wbuf0 | wbuf1 | ibuf0 | ibuf1 ]

  const int tid   = threadIdx.x;
  const int b     = blockIdx.z;
  const int y0    = blockIdx.y * ROWS;
  const int cout0 = blockIdx.x * BM;
  const int wave  = tid >> 5;
  const int lane  = tid & 31;
  const int m0    = wave * 16;
  const int lhalf = lane >> 4;
  const int l16   = lane & 15;

  const __bf16* inb = in + (size_t)b * NPIX * CIN;

  auto stage = [&](int chunk, int bufi) {
    const int cin0 = chunk * KC;
    __bf16* wl = smem + bufi * WELEMS;
    __bf16* il = smem + 2 * WELEMS + bufi * IELEMS;
    // weights: 9*BM rows x 32 bf16 (64B) = 4x16B async units per row
    for (int u = tid; u < 9 * BM * 4; u += 128) {
      int part = u & 3, row = u >> 2;
      int tap = row / BM, co = row - tap * BM;
      cp16_async(&wl[(tap * BM + co) * CSTR + part * 8],
                 wrep + ((size_t)tap * COUT + cout0 + co) * CIN + cin0 + part * 8);
    }
    // input rows y0-1..y0+2, cols -1..64 (zero halo), KC cins per (row,col)
    for (int u = tid; u < INROWS * IN_COLS * 4; u += 128) {
      int part = u & 3;
      int cc = (u >> 2) % IN_COLS;
      int r  = (u >> 2) / IN_COLS;
      int yi = y0 + r - 1;
      int xi = cc - 1;
      __bf16* dst = &il[(r * IN_COLS + cc) * CSTR + part * 8];
      if ((unsigned)yi < HO && (unsigned)xi < WO) {
        cp16_async(dst, inb + ((size_t)yi * WO + xi) * CIN + cin0 + part * 8);
      } else {
        v8bf z = {};
        *(v8bf*)dst = z;
      }
    }
  };

  union V8F  { v8f v;   float f[8]; };
  union V16B { v16bf v; v8bf  h[2]; };

  V8F acc[ROWS][4];
#pragma unroll
  for (int r2 = 0; r2 < ROWS; ++r2)
#pragma unroll
    for (int nt = 0; nt < 4; ++nt)
#pragma unroll
      for (int e = 0; e < 8; ++e) acc[r2][nt].f[e] = 0.f;

  stage(0, 0);
  for (int chunk = 0; chunk < NCHUNK; ++chunk) {
    wait_async_lds();       // our wave's staged data is in LDS
    __syncthreads();        // everyone's is
    if (chunk + 1 < NCHUNK) stage(chunk + 1, (chunk + 1) & 1);  // prefetch

    const int cur = chunk & 1;
    const __bf16* wl = smem + cur * WELEMS;
    const __bf16* il = smem + 2 * WELEMS + cur * IELEMS;
#pragma unroll
    for (int tap = 0; tap < 9; ++tap) {
      const int ky = tap / 3, kx = tap - 3 * ky;
      // A fragment (16x32 bf16): lanes 0-15 -> K0-7/K16-23, lanes 16-31 -> K8-15/K24-31
      V16B a;
      const __bf16* ap = wl + (tap * BM + m0 + l16) * CSTR;
      a.h[0] = *(const v8bf*)(ap + lhalf * 8);
      a.h[1] = *(const v8bf*)(ap + 16 + lhalf * 8);
#pragma unroll
      for (int r2 = 0; r2 < ROWS; ++r2) {
#pragma unroll
        for (int nt = 0; nt < 4; ++nt) {
          int xq = nt * 16 + l16;
          // B fragment (32x16 bf16): lanes 0-15 -> K0-7/K16-23, lanes 16-31 -> K8-15/K24-31
          const __bf16* bp = il + ((r2 + ky) * IN_COLS + xq + kx) * CSTR + lhalf * 8;
          V16B bb;
          bb.h[0] = *(const v8bf*)bp;
          bb.h[1] = *(const v8bf*)(bp + 16);
          acc[r2][nt].v = __builtin_amdgcn_wmma_f32_16x16x32_bf16(
              false, a.v, false, bb.v, (short)0, acc[r2][nt].v, false, false);
        }
      }
    }
  }

  // Epilogue: demodulate, add scaled noise, leaky ReLU, optional next-conv
  // modulation folded into the bf16 NHWC store (16B vector stores).
#pragma unroll
  for (int r2 = 0; r2 < ROWS; ++r2) {
    const int yy = y0 + r2;
    const float* nzrow = noise + (size_t)b * NPIX + yy * WO;
#pragma unroll
    for (int nt = 0; nt < 4; ++nt) {
      int xq = nt * 16 + l16;
      float nv = nzrow[xq];
      union { v8bf v; __bf16 e[8]; } pk;
#pragma unroll
      for (int r = 0; r < 8; ++r) {
        int cout = cout0 + m0 + lhalf * 8 + r;   // C/D layout: M=(lane/16)*8+vgpr, N=lane%16
        float v = acc[r2][nt].f[r] * dmod[b * COUT + cout] + nscale[cout] * nv;
        v = (v >= 0.f) ? v : LRELU * v;
        if (out_f32) out_f32[(((size_t)b * COUT + cout) * HO + yy) * WO + xq] = v;
        if (postsc) v *= postsc[b * COUT + cout];
        pk.e[r] = (__bf16)v;
      }
      if (out_bf16) {
        *(v8bf*)&out_bf16[(((size_t)b * HO + yy) * WO + xq) * CIN + cout0 + m0 + lhalf * 8] = pk.v;
      }
    }
  }
}

// rgb[b,c,p] = sum_k rgbw[c,k]*sr[b,k]*h[b,k,p]   (1x1 conv, no demod)
__global__ void rgb_kernel(const float* __restrict__ h,
                           const float* __restrict__ rgbw,
                           const float* __restrict__ sr,
                           float* __restrict__ out) {
  int p = blockIdx.x * 256 + threadIdx.x;
  int c = blockIdx.y;
  int b = blockIdx.z;
  const float* hb = h + ((size_t)b * COUT) * NPIX + p;
  float a = 0.f;
  for (int k = 0; k < COUT; ++k)
    a += rgbw[c * COUT + k] * sr[b * COUT + k] * hb[(size_t)k * NPIX];
  out[((size_t)b * 3 + c) * NPIX + p] = a;
}

// ---------------------------------------------------------------------------
extern "C" void kernel_launch(void* const* d_in, const int* in_sizes, int n_in,
                              void* d_out, int out_size, void* d_ws, size_t ws_size,
                              hipStream_t stream) {
  (void)in_sizes; (void)n_in; (void)out_size; (void)ws_size;
  const float* x    = (const float*)d_in[0];
  const float* w    = (const float*)d_in[1];
  const float* w1   = (const float*)d_in[2];
  const float* m1w  = (const float*)d_in[3];
  const float* m1b  = (const float*)d_in[4];
  const float* ns1  = (const float*)d_in[5];
  const float* w2   = (const float*)d_in[6];
  const float* m2w  = (const float*)d_in[7];
  const float* m2b  = (const float*)d_in[8];
  const float* ns2  = (const float*)d_in[9];
  const float* wrgb = (const float*)d_in[10];
  const float* mrw  = (const float*)d_in[11];
  const float* mrb  = (const float*)d_in[12];
  const float* n1   = (const float*)d_in[13];
  const float* n2   = (const float*)d_in[14];
  float* out = (float*)d_out;

  size_t off = 0;
  auto alloc = [&](size_t bytes) -> void* {
    void* p = (char*)d_ws + off;
    off += (bytes + 255) & ~(size_t)255;
    return p;
  };
  float*  s1    = (float*)alloc((size_t)B_ * CIN * 4);
  float*  s2    = (float*)alloc((size_t)B_ * COUT * 4);
  float*  sr    = (float*)alloc((size_t)B_ * COUT * 4);
  float*  dm1   = (float*)alloc((size_t)B_ * COUT * 4);
  float*  dm2   = (float*)alloc((size_t)B_ * COUT * 4);
  float*  wsqT1 = (float*)alloc((size_t)COUT * CIN * 4);
  float*  wsqT2 = (float*)alloc((size_t)COUT * CIN * 4);
  __bf16* wrep1 = (__bf16*)alloc((size_t)9 * COUT * CIN * 2);
  __bf16* wrep2 = (__bf16*)alloc((size_t)9 * COUT * CIN * 2);
  __bf16* up    = (__bf16*)alloc((size_t)B_ * CIN * NPIX * 2);   // NHWC
  __bf16* h1    = (__bf16*)alloc((size_t)B_ * COUT * NPIX * 2);  // NHWC, pre-scaled by s2

  dim3 blk256(256), blk128(128);
  style_kernel<<<dim3(CIN / 256, B_), blk256, 0, stream>>>(w, m1w, m1b, s1);
  style_kernel<<<dim3(COUT / 256, B_), blk256, 0, stream>>>(w, m2w, m2b, s2);
  style_kernel<<<dim3(COUT / 256, B_), blk256, 0, stream>>>(w, mrw, mrb, sr);
  wsqT_kernel<<<(COUT * CIN) / 256, blk256, 0, stream>>>(w1, wsqT1);
  wsqT_kernel<<<(COUT * CIN) / 256, blk256, 0, stream>>>(w2, wsqT2);
  demod_kernel<<<dim3(COUT / 256, B_), blk256, 0, stream>>>(s1, wsqT1, dm1);
  demod_kernel<<<dim3(COUT / 256, B_), blk256, 0, stream>>>(s2, wsqT2, dm2);
  repack_kernel<<<(int)(((size_t)9 * COUT * CIN + 255) / 256), blk256, 0, stream>>>(w1, wrep1);
  repack_kernel<<<(int)(((size_t)9 * COUT * CIN + 255) / 256), blk256, 0, stream>>>(w2, wrep2);
  upmod_nhwc_kernel<<<dim3(CIN / 32, HO, B_), blk256, 0, stream>>>(x, s1, up);

  dim3 cgrid(COUT / BM, HO / ROWS, B_);
  // conv1: modulated+upsampled input; epilogue = demod+noise1+lrelu, then *s2 -> h1 (bf16 NHWC)
  conv3x3_wmma<<<cgrid, blk128, LDS_BYTES, stream>>>(
      up, wrep1, dm1, n1, ns1, s2, nullptr, h1);
  // conv2: h1 already carries s2; epilogue = demod+noise2+lrelu -> h (f32 NCHW in d_out)
  conv3x3_wmma<<<cgrid, blk128, LDS_BYTES, stream>>>(
      h1, wrep2, dm2, n2, ns2, nullptr, out, nullptr);
  // toRGB from f32 h
  rgb_kernel<<<dim3(NPIX / 256, 3, B_), blk256, 0, stream>>>(
      out, wrgb, sr, out + (size_t)B_ * COUT * NPIX);
}
